// LocalSparseMultiHeadAttention_38594576122309
// MI455X (gfx1250) — matrix operand order repaired and checked by reading, not compiled, not run on hardware
//
#include <hip/hip_runtime.h>

typedef __attribute__((ext_vector_type(16))) __bf16 v16bf;
typedef __attribute__((ext_vector_type(8)))  __bf16 v8bf;
typedef __attribute__((ext_vector_type(4)))  __bf16 v4bf;
typedef __attribute__((ext_vector_type(8)))  float  v8f;
typedef __attribute__((ext_vector_type(4)))  float  v4f;

// ---------------------------------------------------------------------------
// WMMA helpers (CDNA5 gfx1250, wave32)
// A fragment (16x32 bf16): lane -> M = lane%16; elements 0..7 -> K = half*8+e,
//                          elements 8..15 -> K = 16 + half*8 + (e-8).
// B fragment (32x16 bf16): lane -> N = lane%16; same K mapping as A.
// C/D fragment (16x16 f32): VGPR r, lane -> (M = r + 8*half, N = lane%16).
// ---------------------------------------------------------------------------
__device__ inline v8f wmma_bf16(v16bf a, v16bf b, v8f c) {
  return __builtin_amdgcn_wmma_f32_16x16x32_bf16(false, a, false, b, (short)0, c,
                                                 false, false);
}

__device__ inline v16bf load_fragA(const __bf16* row, int half, int koff) {
  v8bf lo = *(const v8bf*)(row + koff + half * 8);
  v8bf hi = *(const v8bf*)(row + koff + 16 + half * 8);
  v16bf r;
#pragma unroll
  for (int i = 0; i < 8; ++i) { r[i] = lo[i]; r[i + 8] = hi[i]; }
  return r;
}

__device__ inline v16bf load_fragB(const __bf16* row, int half, int koff) {
  v8bf lo = *(const v8bf*)(row + koff + half * 8);
  v8bf hi = *(const v8bf*)(row + koff + 16 + half * 8);
  v16bf r;
#pragma unroll
  for (int i = 0; i < 8; ++i) { r[i] = lo[i]; r[i + 8] = hi[i]; }
  return r;
}

// ---------------------------------------------------------------------------
// gfx1250 async global->LDS copy (ASYNCcnt), per CDNA5 ISA.
// 16B per lane, no VGPR round trip. LDS offset = low 32 bits of generic ptr.
// ---------------------------------------------------------------------------
__device__ inline void async_copy_b128(void* lds_dst, const void* gsrc) {
  unsigned l = (unsigned)(unsigned long long)lds_dst;
  unsigned long long g = (unsigned long long)gsrc;
  asm volatile("global_load_async_to_lds_b128 %0, %1, off"
               :: "v"(l), "v"(g) : "memory");
}
__device__ inline void wait_async() {
  asm volatile("s_wait_asynccnt 0x0" ::: "memory");
}

// ---------------------------------------------------------------------------
// Weight pre-transpose: Wt[n][k] = (bf16)W[k][n].  N fixed at 512.
// Tiled through LDS so both global accesses are coalesced. One-time cost.
// ---------------------------------------------------------------------------
__global__ __launch_bounds__(256) void transpose_w_kernel(
    const float* __restrict__ W, __bf16* __restrict__ Wt, int K) {
  __shared__ float tile[32][33];
  int k0 = blockIdx.x * 32, n0 = blockIdx.y * 32;
  int tx = threadIdx.x & 31, ty = threadIdx.x >> 5;  // 32 x 8
#pragma unroll
  for (int j = 0; j < 32; j += 8)
    tile[ty + j][tx] = W[(size_t)(k0 + ty + j) * 512 + n0 + tx];
  __syncthreads();
#pragma unroll
  for (int j = 0; j < 32; j += 8)
    Wt[(size_t)(n0 + ty + j) * K + k0 + tx] = (__bf16)tile[tx][ty + j];
}

// ---------------------------------------------------------------------------
// GEMM: C[M,512] = A[M,K] * W[K,512] + bias. bf16 WMMA, f32 accumulate.
// K, CONV compile-time: constant-folded addressing, vectorized stagers.
// CONV: A(m, w*512+kk) = q[b, s+w-2, kk] (causal width-3 conv), K = 1536.
// B tiles staged straight from pre-transposed bf16 weights with async-to-LDS.
// Block: 128 threads (4 waves), tile 64x64, K-step 32.
// ---------------------------------------------------------------------------
template <int K, bool CONV>
__global__ __launch_bounds__(128) void gemm_bf16_kernel(
    const float* __restrict__ A, const __bf16* __restrict__ Wt,
    const float* __restrict__ bias, float* __restrict__ Cf,
    __bf16* __restrict__ Cbf) {
  __shared__ __bf16 sA[64 * 40];  // [m][k], pitch 40 -> conflict-free b128 frags
  __shared__ __bf16 sB[64 * 40];  // [n][k] tile of Wt

  const int t = threadIdx.x;
  const int wave = t >> 5, lane = t & 31;
  const int l16 = lane & 15, half = lane >> 4;
  const int m0 = blockIdx.x * 64;
  const int n0 = blockIdx.y * 64;

  v8f acc[4];
#pragma unroll
  for (int nt = 0; nt < 4; ++nt)
#pragma unroll
    for (int i = 0; i < 8; ++i) acc[nt][i] = 0.f;

  for (int kt = 0; kt < (K >> 5); ++kt) {
    // B tile: 64 rows x 32 bf16 from Wt -- pure copy, fire async DMA first.
#pragma unroll
    for (int i = 0; i < 2; ++i) {
      int e = t + i * 128;           // 256 x v8bf
      int nn = e >> 2, c8 = e & 3;
      async_copy_b128(&sB[nn * 40 + c8 * 8],
                      &Wt[(size_t)(n0 + nn) * K + kt * 32 + c8 * 8]);
    }
    // A tile: 64 rows x 32 k, float4 loads + packed bf16 stores.
#pragma unroll
    for (int i = 0; i < 4; ++i) {
      int e = t + i * 128;           // 512 x v4f
      int r = e >> 3, c4 = e & 7;
      v4f v;
      if (CONV) {
        int kg0 = kt * 32;
        int w = kg0 >> 9, kk = (kg0 & 511) + c4 * 4;
        int m = m0 + r;
        int b = m >> 11, s = m & 2047;
        int ss = s + w - 2;
        if (ss >= 0) {
          v = *(const v4f*)&A[((size_t)(b * 2048 + ss)) * 512 + kk];
        } else {
          v = (v4f){0.f, 0.f, 0.f, 0.f};
        }
      } else {
        v = *(const v4f*)&A[(size_t)(m0 + r) * K + kt * 32 + c4 * 4];
      }
      v4bf pk;
      pk[0] = (__bf16)v.x; pk[1] = (__bf16)v.y;
      pk[2] = (__bf16)v.z; pk[3] = (__bf16)v.w;
      *(v4bf*)&sA[r * 40 + c4 * 4] = pk;
    }
    wait_async();
    __syncthreads();

    v16bf af = load_fragA(&sA[(wave * 16 + l16) * 40], half, 0);
#pragma unroll
    for (int nt = 0; nt < 4; ++nt) {
      v16bf bf = load_fragB(&sB[(nt * 16 + l16) * 40], half, 0);
      acc[nt] = wmma_bf16(af, bf, acc[nt]);
    }
    __syncthreads();
  }

#pragma unroll
  for (int nt = 0; nt < 4; ++nt) {
    int col = n0 + nt * 16 + l16;
    float bv = bias[col];
#pragma unroll
    for (int r = 0; r < 8; ++r) {
      int row = m0 + wave * 16 + r + 8 * half;
      float v = acc[nt][r] + bv;
      Cf[(size_t)row * 512 + col] = v;
      if (Cbf) Cbf[(size_t)row * 512 + col] = (__bf16)v;
    }
  }
}

// ---------------------------------------------------------------------------
// Mask statistics: Eg[b][k] = exp(mask*-1e9); Tin[b][n], Gmax[b][n], Tb[b].
// ---------------------------------------------------------------------------
__global__ void mask_prep_kernel(const float* __restrict__ mask, float* Eg,
                                 float* Tin, float* Tb, float* Gmax) {
  int b = blockIdx.x, t = threadIdx.x;
  __shared__ float rs[256], rm[256];
  float tot = 0.f;
  for (int n = 0; n < 8; ++n) {
    int k = n * 256 + t;
    float g = mask[b * 2048 + k] * -1e9f;
    float e = __expf(g);
    Eg[b * 2048 + k] = e;
    rs[t] = e; rm[t] = g;
    __syncthreads();
    for (int s = 128; s > 0; s >>= 1) {
      if (t < s) { rs[t] += rs[t + s]; rm[t] = fmaxf(rm[t], rm[t + s]); }
      __syncthreads();
    }
    if (t == 0) { Tin[b * 8 + n] = rs[0]; Gmax[b * 8 + n] = rm[0]; tot += rs[0]; }
    __syncthreads();
  }
  if (t == 0) Tb[b] = tot;
}

// Uin[b][h][n][d] = sum over block-n keys of Eg[k]*vc[k][h*64+d]
__global__ __launch_bounds__(512) void uin_kernel(const float* __restrict__ Eg,
                                                  const float* __restrict__ vc,
                                                  float* __restrict__ Uin) {
  int bid = blockIdx.x;
  int b = bid >> 3, n = bid & 7;
  int j = threadIdx.x;  // h*64 + d
  float acc = 0.f;
  for (int kk = 0; kk < 256; ++kk) {
    int key = b * 2048 + n * 256 + kk;
    acc += Eg[key] * vc[(size_t)key * 512 + j];
  }
  int h = j >> 6, d = j & 63;
  Uin[((b * 8 + h) * 8 + n) * 64 + d] = acc;
}

__global__ void u_kernel(const float* __restrict__ Uin, float* __restrict__ U) {
  int idx = blockIdx.x * 256 + threadIdx.x;  // B*H*64 = 1024
  if (idx >= 2 * 8 * 64) return;
  int bh = idx >> 6, d = idx & 63;
  float s = 0.f;
  for (int n = 0; n < 8; ++n) s += Uin[(bh * 8 + n) * 64 + d];
  U[idx] = s;
}

// ---------------------------------------------------------------------------
// Fused block attention. Grid: B*H*nb*4 blocks; each block = 64 query rows of
// one (b,h,n) diagonal block. 128 threads = 4 waves, each wave owns 16 rows.
// Q/K tiles staged with gfx1250 async global->LDS DMA.
// ---------------------------------------------------------------------------
__global__ __launch_bounds__(128) void attn_kernel(
    const __bf16* __restrict__ qcb, const __bf16* __restrict__ kcb,
    const float* __restrict__ vc, const float* __restrict__ mask,
    const float* __restrict__ Tin, const float* __restrict__ Tb,
    const float* __restrict__ Gmax, const float* __restrict__ Uin,
    const float* __restrict__ U, float* __restrict__ attn,
    float* __restrict__ rowcoeff, float* __restrict__ concat) {
  extern __shared__ char smem[];
  __bf16* sQ  = (__bf16*)smem;             // 64 x 72  (phase 1)
  __bf16* sK  = (__bf16*)(smem + 9216);    // 256 x 72 (phase 1)
  __bf16* sP  = (__bf16*)smem;             // 64 x 264 (phase 2)
  __bf16* sVT = (__bf16*)(smem + 33792);   // 64 x 264 (phase 2)

  int bid = blockIdx.x;
  int qt = bid & 3, n = (bid >> 2) & 7, h = (bid >> 5) & 7, b = bid >> 8;
  int t = threadIdx.x;
  int wave = t >> 5, lane = t & 31, l16 = lane & 15, half = lane >> 4;
  int qbase = n * 256 + qt * 64;

  // Stage Q (64x64) and K (256x64) bf16 head slices: async DMA to LDS.
#pragma unroll
  for (int i = 0; i < 4; ++i) {
    int e = t + i * 128;
    int r = e >> 3, c8 = e & 7;
    async_copy_b128(&sQ[r * 72 + c8 * 8],
                    &qcb[((size_t)(b * 2048 + qbase + r)) * 512 + h * 64 + c8 * 8]);
  }
#pragma unroll
  for (int i = 0; i < 16; ++i) {
    int e = t + i * 128;
    int r = e >> 3, c8 = e & 7;
    async_copy_b128(&sK[r * 72 + c8 * 8],
                    &kcb[((size_t)(b * 2048 + n * 256 + r)) * 512 + h * 64 + c8 * 8]);
  }
  wait_async();
  __syncthreads();

  // QK^T: wave's 16 rows x 256 keys, depth 64 in two 32-K steps.
  v8f acc[16];
#pragma unroll
  for (int nt = 0; nt < 16; ++nt)
#pragma unroll
    for (int i = 0; i < 8; ++i) acc[nt][i] = 0.f;
#pragma unroll
  for (int ks = 0; ks < 2; ++ks) {
    v16bf af = load_fragA(&sQ[(wave * 16 + l16) * 72], half, ks * 32);
#pragma unroll
    for (int nt = 0; nt < 16; ++nt) {
      v16bf bf = load_fragB(&sK[(nt * 16 + l16) * 72], half, ks * 32);
      acc[nt] = wmma_bf16(af, bf, acc[nt]);
    }
  }

  // Per-lane mask logits for its 16 key columns.
  float gk[16];
#pragma unroll
  for (int nt = 0; nt < 16; ++nt)
    gk[nt] = mask[b * 2048 + n * 256 + nt * 16 + l16] * -1e9f;

  float goff = -3.4e38f;
  for (int nn = 0; nn < 8; ++nn)
    if (nn != n) goff = fmaxf(goff, Gmax[b * 8 + nn]);
  float Toff = Tb[b] - Tin[b * 8 + n];

  // Full-row softmax stats in registers: each row lives in one 16-lane group,
  // so 4 xor-shuffles reduce it (wave32 friendly).
  const float scale = 0.125f;  // 1/sqrt(64)
  float Z[8], enm[8], cf[8];
#pragma unroll
  for (int r = 0; r < 8; ++r) {
    float mx = goff;
#pragma unroll
    for (int nt = 0; nt < 16; ++nt) mx = fmaxf(mx, acc[nt][r] * scale + gk[nt]);
    for (int s = 1; s < 16; s <<= 1) mx = fmaxf(mx, __shfl_xor(mx, s, 32));
    float sum = 0.f;
#pragma unroll
    for (int nt = 0; nt < 16; ++nt) {
      float p = __expf(acc[nt][r] * scale + gk[nt] - mx);
      acc[nt][r] = p;  // keep unnormalized probs in place
      sum += p;
    }
    for (int s = 1; s < 16; s <<= 1) sum += __shfl_xor(sum, s, 32);
    enm[r] = __expf(-mx);
    Z[r] = sum + enm[r] * Toff;
    cf[r] = enm[r] / Z[r];
  }
  __syncthreads();  // done with sQ/sK; smem re-used below

  // Write normalized in-block attn (non-temporal: write-once 268MB stream,
  // don't evict the 192MB L2) + row coeff; spill P~ (bf16) to LDS.
#pragma unroll
  for (int r = 0; r < 8; ++r) {
    int lr = wave * 16 + r + 8 * half;
    int qg = qbase + lr;
    float invZ = 1.f / Z[r];
    float* arow = attn + ((size_t)((b * 8 + h) * 2048 + qg)) * 2048 + n * 256;
#pragma unroll
    for (int nt = 0; nt < 16; ++nt) {
      float p = acc[nt][r];
      __builtin_nontemporal_store(p * invZ, &arow[nt * 16 + l16]);
      sP[lr * 264 + nt * 16 + l16] = (__bf16)p;
    }
    if (l16 == 0) rowcoeff[(b * 8 + h) * 2048 + qg] = cf[r];
  }

  // Build V^T tile [d][key] in LDS (f32 -> bf16).
#pragma unroll
  for (int i = 0; i < 32; ++i) {
    int e = t + i * 128;
    int key = e >> 4, d4 = e & 15;
    const v4f v =
        *(const v4f*)&vc[((size_t)(b * 2048 + n * 256 + key)) * 512 + h * 64 + d4 * 4];
    sVT[(d4 * 4 + 0) * 264 + key] = (__bf16)v.x;
    sVT[(d4 * 4 + 1) * 264 + key] = (__bf16)v.y;
    sVT[(d4 * 4 + 2) * 264 + key] = (__bf16)v.z;
    sVT[(d4 * 4 + 3) * 264 + key] = (__bf16)v.w;
  }
  __syncthreads();

  // P~ @ V_blk : 16 rows x 64 d, 256 keys in eight 32-K steps.
  v8f oacc[4];
#pragma unroll
  for (int dt = 0; dt < 4; ++dt)
#pragma unroll
    for (int i = 0; i < 8; ++i) oacc[dt][i] = 0.f;
#pragma unroll
  for (int ks = 0; ks < 8; ++ks) {
    v16bf af = load_fragA(&sP[(wave * 16 + l16) * 264], half, ks * 32);
#pragma unroll
    for (int dt = 0; dt < 4; ++dt) {
      v16bf bf = load_fragB(&sVT[(dt * 16 + l16) * 264], half, ks * 32);
      oacc[dt] = wmma_bf16(af, bf, oacc[dt]);
    }
  }

  // out = (P~ V + e^{-m} (U - Uin)) / Z  ->  concat[b][q][h*64+d]
#pragma unroll
  for (int dt = 0; dt < 4; ++dt) {
    int d = dt * 16 + l16;
    float u = U[(b * 8 + h) * 64 + d];
    float ui = Uin[((b * 8 + h) * 8 + n) * 64 + d];
    float du = u - ui;
#pragma unroll
    for (int r = 0; r < 8; ++r) {
      int qg = qbase + wave * 16 + r + 8 * half;
      float val = (oacc[dt][r] + enm[r] * du) / Z[r];
      concat[((size_t)(b * 2048 + qg)) * 512 + h * 64 + d] = val;
    }
  }
}

// ---------------------------------------------------------------------------
// Off-block rank-1 fill: attn[row][k] = coeff(row) * Eg[b][k]  (k not in blk).
// Dominant ~260MB stream: v4f non-temporal stores (block edges are
// 256-aligned, so a float4 never straddles the in-block hole).
// ---------------------------------------------------------------------------
__global__ __launch_bounds__(256) void offfill_kernel(
    const float* __restrict__ rowcoeff, const float* __restrict__ Eg,
    float* __restrict__ attn) {
  int row = blockIdx.x;  // (b*8+h)*2048 + q, total 32768
  int t = threadIdx.x;
  int q = row & 2047;
  int b = row >> 14;
  float c = rowcoeff[row];
  const v4f* eg4 = (const v4f*)(Eg + b * 2048);
  v4f* arow4 = (v4f*)(attn + (size_t)row * 2048);
  int nblk = q >> 8;
#pragma unroll
  for (int it = 0; it < 2; ++it) {
    int k4 = it * 256 + t;             // 512 v4f per row
    if ((k4 >> 6) != nblk) {
      v4f e = eg4[k4];
      v4f o = e * c;
      __builtin_nontemporal_store(o, &arow4[k4]);
    }
  }
}

// ---------------------------------------------------------------------------
extern "C" void kernel_launch(void* const* d_in, const int* in_sizes, int n_in,
                              void* d_out, int out_size, void* d_ws, size_t ws_size,
                              hipStream_t stream) {
  (void)in_sizes; (void)n_in; (void)out_size; (void)ws_size;

  const float* v    = (const float*)d_in[0];
  const float* k    = (const float*)d_in[1];
  const float* q    = (const float*)d_in[2];
  const float* mask = (const float*)d_in[3];
  const float* wq   = (const float*)d_in[4];   // [1536,512] flat
  const float* wqb  = (const float*)d_in[5];
  const float* wk   = (const float*)d_in[6];   // [512,512]
  const float* wkb  = (const float*)d_in[7];
  const float* wv   = (const float*)d_in[8];
  const float* wvb  = (const float*)d_in[9];
  const float* dw   = (const float*)d_in[10];
  const float* db   = (const float*)d_in[11];

  char* ws = (char*)d_ws;
  size_t o = 0;
  auto take = [&](size_t bytes) -> char* {
    char* p = ws + o;
    o += (bytes + 255) & ~(size_t)255;
    return p;
  };
  float*  qc     = (float*)take(4096ull * 512 * 4);
  float*  kc     = (float*)take(4096ull * 512 * 4);
  float*  vc     = (float*)take(4096ull * 512 * 4);
  __bf16* qcb    = (__bf16*)take(4096ull * 512 * 2);
  __bf16* kcb    = (__bf16*)take(4096ull * 512 * 2);
  float*  concat = (float*)take(4096ull * 512 * 4);
  __bf16* wkT    = (__bf16*)take(512ull * 512 * 2);
  __bf16* wvT    = (__bf16*)take(512ull * 512 * 2);
  __bf16* wqT    = (__bf16*)take(512ull * 1536 * 2);
  __bf16* dwT    = (__bf16*)take(512ull * 512 * 2);
  float*  Eg     = (float*)take(2ull * 2048 * 4);
  float*  Tin    = (float*)take(16 * 4);
  float*  Tb     = (float*)take(2 * 4);
  float*  Gm     = (float*)take(16 * 4);
  float*  Uin    = (float*)take(2ull * 8 * 8 * 64 * 4);
  float*  U      = (float*)take(2ull * 8 * 64 * 4);
  float*  rc     = (float*)take(2ull * 8 * 2048 * 4);

  float* out  = (float*)d_out;                 // [2,2048,512]
  float* attn = out + 2ull * 2048 * 512;       // [2,8,2048,2048]

  // One-time bf16 weight transposes (coalesced, trivial vs. 340MB total traffic)
  transpose_w_kernel<<<dim3(16, 16), 256, 0, stream>>>(wk, wkT, 512);
  transpose_w_kernel<<<dim3(16, 16), 256, 0, stream>>>(wv, wvT, 512);
  transpose_w_kernel<<<dim3(48, 16), 256, 0, stream>>>(wq, wqT, 1536);
  transpose_w_kernel<<<dim3(16, 16), 256, 0, stream>>>(dw, dwT, 512);

  dim3 gg(64, 8);
  gemm_bf16_kernel<512, false><<<gg, 128, 0, stream>>>(k, wkT, wkb, kc, kcb);
  gemm_bf16_kernel<512, false><<<gg, 128, 0, stream>>>(v, wvT, wvb, vc, nullptr);
  gemm_bf16_kernel<1536, true><<<gg, 128, 0, stream>>>(q, wqT, wqb, qc, qcb);
  mask_prep_kernel<<<2, 256, 0, stream>>>(mask, Eg, Tin, Tb, Gm);
  uin_kernel<<<16, 512, 0, stream>>>(Eg, vc, Uin);
  u_kernel<<<4, 256, 0, stream>>>(Uin, U);
  attn_kernel<<<512, 128, 67584, stream>>>(qcb, kcb, vc, mask, Tin, Tb, Gm, Uin, U,
                                           attn, rc, concat);
  offfill_kernel<<<32768, 256, 0, stream>>>(rc, Eg, attn);
  gemm_bf16_kernel<512, false><<<gg, 128, 0, stream>>>(concat, dwT, db, out, nullptr);
}
